// TGC_LSTM_61272003445086
// MI455X (gfx1250) — matrix operand order repaired and checked by reading, not compiled, not run on hardware
//
#include <hip/hip_runtime.h>
#include <hip/hip_bf16.h>

// ---------------------------------------------------------------------------
// TGC-LSTM on MI455X (gfx1250):
//   comb = A @ (X@gcWx + H@gcWh)  +  (X@liWx + H@liWh + sum_of_biases)
// Big dense GEMM [4096,4096]x[4096,256] via V_WMMA_F32_16X16X32_BF16 with
// double-buffered fragment loads; A normalized once to bf16 (32MB, L2-resident).
// ---------------------------------------------------------------------------

#define NN   4096
#define HH   64
#define GW   256      // 4*H gate width
#define TT   48

typedef __attribute__((ext_vector_type(16))) __bf16 v16bf;
typedef __attribute__((ext_vector_type(8)))  float  v8f;

// ---------------- A normalization: dinv = rsqrt(rowsum(adj)+1) --------------
__global__ void __launch_bounds__(256)
degree_kernel(const float* __restrict__ adj, float* __restrict__ dinv) {
    __shared__ float red[256];
    const int row = blockIdx.x;
    float s = 0.f;
    for (int j = threadIdx.x; j < NN; j += 256)
        s += adj[(size_t)row * NN + j];
    red[threadIdx.x] = s;
    __syncthreads();
    for (int st = 128; st > 0; st >>= 1) {
        if (threadIdx.x < st) red[threadIdx.x] += red[threadIdx.x + st];
        __syncthreads();
    }
    if (threadIdx.x == 0) dinv[row] = rsqrtf(red[0] + 1.0f);  // +I guarantees >0
}

// A_bf16[i][j] = bf16( dinv[i] * (adj[i][j] + (i==j)) * dinv[j] ), row-major
__global__ void __launch_bounds__(256)
normalize_kernel(const float* __restrict__ adj, const float* __restrict__ dinv,
                 __hip_bfloat16* __restrict__ Abf) {
    const int i = blockIdx.x;
    const float di = dinv[i];
    for (int j = threadIdx.x; j < NN; j += 256) {
        float a = adj[(size_t)i * NN + j] + (i == j ? 1.0f : 0.0f);
        Abf[(size_t)i * NN + j] = __float2bfloat16(di * a * dinv[j]);
    }
}

__global__ void __launch_bounds__(256)
zero_kernel(float* __restrict__ p, int n) {
    int i = blockIdx.x * 256 + threadIdx.x;
    if (i < n) p[i] = 0.0f;
}

// ------------- per-cell pre-pass: U (bf16, transposed) and V (f32) ----------
// U[n][c] = X[n]@gcWx[:,c] + H[n]@gcWh[:,c]           -> Ut[c*4096+n] (bf16)
// V[n][c] = X[n]@liWx[:,c] + H[n]@liWh[:,c] + biases  -> V[n*256+c]   (f32)
__global__ void __launch_bounds__(256)
cell_pre_kernel(const float* __restrict__ X, int Din,
                const float* __restrict__ Hprev,
                const float* __restrict__ gcWx, const float* __restrict__ gcWh,
                const float* __restrict__ liWx, const float* __restrict__ liWh,
                const float* __restrict__ b0, const float* __restrict__ b1,
                const float* __restrict__ b2, const float* __restrict__ b3,
                __hip_bfloat16* __restrict__ Ut, float* __restrict__ V) {
    __shared__ float xs[64];
    __shared__ float hs[HH];
    const int n = blockIdx.x;
    const int c = threadIdx.x;            // 0..255 gate column
    if (c < Din) xs[c] = X[(size_t)n * Din + c];
    if (c < HH)  hs[c] = Hprev[(size_t)n * HH + c];
    __syncthreads();
    float u = 0.f;
    float vv = b0[c] + b1[c] + b2[c] + b3[c];
    for (int k = 0; k < Din; ++k) {
        u  += xs[k] * gcWx[k * GW + c];
        vv += xs[k] * liWx[k * GW + c];
    }
    #pragma unroll 8
    for (int k = 0; k < HH; ++k) {
        u  += hs[k] * gcWh[k * GW + c];
        vv += hs[k] * liWh[k * GW + c];
    }
    Ut[(size_t)c * NN + n] = __float2bfloat16(u);
    V[(size_t)n * GW + c]  = vv;
}

// ------------------- WMMA fragment helpers ---------------------------------
union Frag { unsigned int u[8]; v16bf v; };

// 16-bit A 16x32 layout: lane holds row m; VGPR i holds K = (i>>2)*16 + half*8 + (i&3)*2 (+1)
__device__ __forceinline__ void load_afrag(Frag& f, const unsigned short* __restrict__ row,
                                           int k, int half) {
    #pragma unroll
    for (int i = 0; i < 8; ++i) {
        const int kA = k + ((i >> 2) << 4) + (half << 3) + ((i & 3) << 1);
        f.u[i] = *reinterpret_cast<const unsigned int*>(row + kA);
    }
}

// 16-bit B 32x16 layout: lane holds col n; VGPR i holds K = (i>>2)*16 + half*8 + (i&3)*2 (+1)
__device__ __forceinline__ void load_bfrag(Frag& f, const unsigned short* __restrict__ row,
                                           int k, int half) {
    #pragma unroll
    for (int i = 0; i < 8; ++i) {
        const int kB = k + ((i >> 2) << 4) + (half << 3) + ((i & 3) << 1);
        f.u[i] = *reinterpret_cast<const unsigned int*>(row + kB);
    }
}

#define WMMA_BF16(ACC, AF, BF) \
    ACC = __builtin_amdgcn_wmma_f32_16x16x32_bf16(false, (AF).v, false, (BF).v, \
                                                  (short)0, ACC, false, false)

// ------------------- WMMA SpMM: comb = A_bf16 @ U + V -----------------------
// Grid: 128 blocks (32-row M strips) x 256 threads (8 waves).
// Wave w owns cols [w*32, w*32+32): 2 M-tiles x 2 N-tiles = 4 accumulators.
// K-loop double-buffered: fragments for tile k+32 load while tile k MMAs run.
__global__ void __launch_bounds__(256)
spmm_wmma_kernel(const __hip_bfloat16* __restrict__ Abf,   // [4096,4096] row-major
                 const __hip_bfloat16* __restrict__ Ut,    // [256,4096]  (U^T)
                 const float* __restrict__ V,              // [4096,256]
                 float* __restrict__ comb) {               // [4096,256]
    const int lane  = threadIdx.x & 31;
    const int wave  = threadIdx.x >> 5;      // 0..7
    const int mBase = blockIdx.x << 5;       // 32-row strip (2 M-tiles)
    const int nBase = wave << 5;             // 32-col strip (2 N-tiles)
    const int half  = lane >> 4;             // 0/1
    const int mrow  = lane & 15;
    const int ncol  = lane & 15;

    const unsigned short* A  = reinterpret_cast<const unsigned short*>(Abf);
    const unsigned short* U  = reinterpret_cast<const unsigned short*>(Ut);
    const unsigned short* ar0 = A + (size_t)(mBase + mrow) * NN;        // M-tile 0
    const unsigned short* ar1 = A + (size_t)(mBase + 16 + mrow) * NN;   // M-tile 1
    const unsigned short* br0 = U + (size_t)(nBase + ncol) * NN;        // N-tile 0
    const unsigned short* br1 = U + (size_t)(nBase + 16 + ncol) * NN;   // N-tile 1

    v8f acc00 = {};   // (m0, n0)
    v8f acc01 = {};   // (m0, n1)
    v8f acc10 = {};   // (m1, n0)
    v8f acc11 = {};   // (m1, n1)

    Frag a0p, a1p, b0p, b1p;   // ping
    Frag a0q, a1q, b0q, b1q;   // pong

    load_afrag(a0p, ar0, 0, half);
    load_afrag(a1p, ar1, 0, half);
    load_bfrag(b0p, br0, 0, half);
    load_bfrag(b1p, br1, 0, half);

    for (int k = 0; k < NN; k += 64) {
        // prefetch A two tiles ahead (global_prefetch_b8)
        const int kpf = (k + 128 < NN) ? k + 128 : 0;
        __builtin_prefetch(ar0 + kpf, 0, 3);
        __builtin_prefetch(ar1 + kpf, 0, 3);

        // stage tile k+32 while MMAs consume tile k
        load_afrag(a0q, ar0, k + 32, half);
        load_afrag(a1q, ar1, k + 32, half);
        load_bfrag(b0q, br0, k + 32, half);
        load_bfrag(b1q, br1, k + 32, half);

        WMMA_BF16(acc00, a0p, b0p);
        WMMA_BF16(acc01, a0p, b1p);
        WMMA_BF16(acc10, a1p, b0p);
        WMMA_BF16(acc11, a1p, b1p);

        // stage tile k+64 (wrap to 0 on the tail; branch-free, result unused)
        const int kn = (k + 64 < NN) ? k + 64 : 0;
        load_afrag(a0p, ar0, kn, half);
        load_afrag(a1p, ar1, kn, half);
        load_bfrag(b0p, br0, kn, half);
        load_bfrag(b1p, br1, kn, half);

        WMMA_BF16(acc00, a0q, b0q);
        WMMA_BF16(acc01, a0q, b1q);
        WMMA_BF16(acc10, a1q, b0q);
        WMMA_BF16(acc11, a1q, b1q);
    }

    // C/D layout: VGPR v -> row = base + v + 8*half, col = nBase + (lane&15)
    #pragma unroll
    for (int v = 0; v < 8; ++v) {
        const int r0 = mBase + v + (half << 3);
        const int r1 = r0 + 16;
        const int c0 = nBase + ncol;
        const int c1 = c0 + 16;
        comb[(size_t)r0 * GW + c0] = acc00[v] + V[(size_t)r0 * GW + c0];
        comb[(size_t)r0 * GW + c1] = acc01[v] + V[(size_t)r0 * GW + c1];
        comb[(size_t)r1 * GW + c0] = acc10[v] + V[(size_t)r1 * GW + c0];
        comb[(size_t)r1 * GW + c1] = acc11[v] + V[(size_t)r1 * GW + c1];
    }
}

// --------------------------- LSTM gate update -------------------------------
__global__ void __launch_bounds__(256)
gates_kernel(const float* __restrict__ comb,
             float* __restrict__ cst, float* __restrict__ hst) {
    const int idx = blockIdx.x * 256 + threadIdx.x;   // 4096*64 threads
    const int n = idx >> 6;
    const int j = idx & 63;
    const float* row = comb + (size_t)n * GW;
    const float ig = row[j];
    const float fg = row[64 + j];
    const float og = row[128 + j];
    const float gg = row[192 + j];
    const float si = 1.0f / (1.0f + __expf(-ig));
    const float sf = 1.0f / (1.0f + __expf(-fg));
    const float so = 1.0f / (1.0f + __expf(-og));
    const float cn = sf * cst[idx] + si * tanhf(gg);
    cst[idx] = cn;
    hst[idx] = so * tanhf(cn);
}

// ------------------------ output projection [N,P=12] ------------------------
__global__ void __launch_bounds__(64)
out_kernel(const float* __restrict__ h1, const float* __restrict__ outW,
           const float* __restrict__ outb, float* __restrict__ out) {
    __shared__ float hl[HH];
    const int n = blockIdx.x;
    const int t = threadIdx.x;
    hl[t] = h1[(size_t)n * HH + t];
    __syncthreads();
    if (t < 12) {
        float s = outb[t];
        #pragma unroll
        for (int k = 0; k < HH; ++k) s += hl[k] * outW[k * 12 + t];
        out[(size_t)n * 12 + t] = s;
    }
}

// ---------------------------------------------------------------------------
extern "C" void kernel_launch(void* const* d_in, const int* in_sizes, int n_in,
                              void* d_out, int out_size, void* d_ws, size_t ws_size,
                              hipStream_t stream) {
    (void)in_sizes; (void)n_in; (void)out_size; (void)ws_size;
    const float* x     = (const float*)d_in[0];   // [1,48,4096*2]
    const float* adj   = (const float*)d_in[1];   // [4096,4096]
    const float* gcWi0 = (const float*)d_in[2];
    const float* gcbi0 = (const float*)d_in[3];
    const float* gcWh0 = (const float*)d_in[4];
    const float* gcbh0 = (const float*)d_in[5];
    const float* liWi0 = (const float*)d_in[6];
    const float* libi0 = (const float*)d_in[7];
    const float* liWh0 = (const float*)d_in[8];
    const float* libh0 = (const float*)d_in[9];
    const float* gcWi1 = (const float*)d_in[10];
    const float* gcbi1 = (const float*)d_in[11];
    const float* gcWh1 = (const float*)d_in[12];
    const float* gcbh1 = (const float*)d_in[13];
    const float* liWi1 = (const float*)d_in[14];
    const float* libi1 = (const float*)d_in[15];
    const float* liWh1 = (const float*)d_in[16];
    const float* libh1 = (const float*)d_in[17];
    const float* outW  = (const float*)d_in[18];
    const float* outb  = (const float*)d_in[19];
    float* out = (float*)d_out;

    // Workspace carve-out (~46 MB)
    char* ws = (char*)d_ws;
    size_t off = 0;
    auto carve = [&](size_t bytes) {
        void* p = ws + off;
        off = (off + bytes + 255) & ~(size_t)255;
        return p;
    };
    __hip_bfloat16* Abf = (__hip_bfloat16*)carve((size_t)NN * NN * 2);  // 32 MB
    __hip_bfloat16* Ut  = (__hip_bfloat16*)carve((size_t)GW * NN * 2);  //  2 MB
    float* V    = (float*)carve((size_t)NN * GW * 4);                   //  4 MB
    float* comb = (float*)carve((size_t)NN * GW * 4);                   //  4 MB
    float* dinv = (float*)carve((size_t)NN * 4);
    float* st   = (float*)carve((size_t)4 * NN * HH * 4);               //  4 MB
    float* h0 = st;
    float* c0 = st + (size_t)NN * HH;
    float* h1 = st + (size_t)2 * NN * HH;
    float* c1 = st + (size_t)3 * NN * HH;

    // One-time (per call) prep
    degree_kernel<<<NN, 256, 0, stream>>>(adj, dinv);
    normalize_kernel<<<NN, 256, 0, stream>>>(adj, dinv, Abf);
    {
        const int nz = 4 * NN * HH;
        zero_kernel<<<(nz + 255) / 256, 256, 0, stream>>>(st, nz);
    }

    // Recurrence
    for (int t = 0; t < TT; ++t) {
        const float* xt = x + (size_t)t * NN * 2;   // [4096,2] row-major, B=1
        // ---- cell 0 ----
        cell_pre_kernel<<<NN, 256, 0, stream>>>(xt, 2, h0,
            gcWi0, gcWh0, liWi0, liWh0, gcbi0, gcbh0, libi0, libh0, Ut, V);
        spmm_wmma_kernel<<<NN / 32, 256, 0, stream>>>(Abf, Ut, V, comb);
        gates_kernel<<<(NN * HH) / 256, 256, 0, stream>>>(comb, c0, h0);
        // ---- cell 1 ----
        cell_pre_kernel<<<NN, 256, 0, stream>>>(h0, HH, h1,
            gcWi1, gcWh1, liWi1, liWh1, gcbi1, gcbh1, libi1, libh1, Ut, V);
        spmm_wmma_kernel<<<NN / 32, 256, 0, stream>>>(Abf, Ut, V, comb);
        gates_kernel<<<(NN * HH) / 256, 256, 0, stream>>>(comb, c1, h1);
    }

    out_kernel<<<NN, 64, 0, stream>>>(h1, outW, outb, out);
}
